// IPEXGPTJBlock_14783277433033
// MI455X (gfx1250) — matrix operand order repaired and checked by reading, not compiled, not run on hardware
//
#include <hip/hip_runtime.h>
#include <stdint.h>

#define Bq 2
#define Sq 2048
#define Dq 4096
#define Hq 16
#define DHq 256
#define FFq 16384

typedef _Float16 ft16;
typedef __attribute__((ext_vector_type(16))) _Float16 v16h;
typedef __attribute__((ext_vector_type(8)))  _Float16 v8h;
typedef __attribute__((ext_vector_type(8)))  float    v8f;

__device__ __forceinline__ v16h cat16(v8h lo, v8h hi) {
  return __builtin_shufflevector(lo, hi, 0,1,2,3,4,5,6,7,8,9,10,11,12,13,14,15);
}
__device__ __forceinline__ v8f wmma_f16(v16h a, v16h b, v8f c) {
  return __builtin_amdgcn_wmma_f32_16x16x32_f16(false, a, false, b, (short)0, c, false, false);
}
__device__ __forceinline__ float gelu_tanh(float x) {
  return 0.5f * x * (1.0f + tanhf(0.7978845608f * (x + 0.044715f * x * x * x)));
}

// ---------------- weight transpose + f32->f16 convert -------------------
// W: K x N (row-major fp32)  ->  WT: N x K (row-major f16)
__global__ __launch_bounds__(256)
void transpose_cvt(const float* __restrict__ W, ft16* __restrict__ WT, int K, int N) {
  __shared__ float tile[32][33];
  const int tx = threadIdx.x & 31;
  const int ty = threadIdx.x >> 5;               // 0..7
  const int n0 = blockIdx.x * 32;
  const int k0 = blockIdx.y * 32;
  for (int i = ty; i < 32; i += 8)
    tile[i][tx] = W[(size_t)(k0 + i) * N + (n0 + tx)];
  __syncthreads();
  for (int i = ty; i < 32; i += 8)
    WT[(size_t)(n0 + i) * K + (k0 + tx)] = (ft16)tile[tx][i];
}

// ---------------- LayerNorm: fp32 in -> f16 out -------------------------
__global__ __launch_bounds__(256)
void ln_kernel(const float* __restrict__ X, const float* __restrict__ W,
               const float* __restrict__ Bb, ft16* __restrict__ Out) {
  __shared__ float red[256];
  const int row = blockIdx.x;
  const float* xr = X + (size_t)row * Dq;
  float s = 0.f;
  for (int j = threadIdx.x; j < Dq; j += 256) s += xr[j];
  red[threadIdx.x] = s; __syncthreads();
  for (int st = 128; st > 0; st >>= 1) {
    if ((int)threadIdx.x < st) red[threadIdx.x] += red[threadIdx.x + st];
    __syncthreads();
  }
  float mu = red[0] * (1.0f / Dq);
  __syncthreads();
  float vs = 0.f;
  for (int j = threadIdx.x; j < Dq; j += 256) { float d = xr[j] - mu; vs += d * d; }
  red[threadIdx.x] = vs; __syncthreads();
  for (int st = 128; st > 0; st >>= 1) {
    if ((int)threadIdx.x < st) red[threadIdx.x] += red[threadIdx.x + st];
    __syncthreads();
  }
  float rstd = rsqrtf(red[0] * (1.0f / Dq) + 1e-5f);
  ft16* orow = Out + (size_t)row * Dq;
  for (int j = threadIdx.x; j < Dq; j += 256)
    orow[j] = (ft16)((xr[j] - mu) * rstd * W[j] + Bb[j]);
}

// ---------------- register-blocked, double-buffered WMMA GEMM -----------
// C[M,N] = A[M,K](f16, row-major) x BT[N,K](f16, row-major = B transposed).
// Block = 8 waves covering 128x128; each wave owns 32 rows x 64 cols
// (2x4 tiles of 16x16) -> 8 wmma per 32-wide k-step. Fragments for k-step
// kb+32 are issued before the WMMAs consuming k-step kb (double buffer).
// OUTMODE: 0 = fp32 (optional +=), 1 = f16, 2 = f16 written per-head
// transposed [B,H,DH,S] (for the V projection feeding flash attention).
template<bool ADD, bool GELU, bool BIAS, int OUTMODE>
__global__ __launch_bounds__(256)
void gemm_wmma(const ft16* __restrict__ A, const ft16* __restrict__ BT,
               const float* __restrict__ bias, void* __restrict__ Cout,
               int M, int N, int K) {
  const int lane = threadIdx.x & 31;
  const int wave = threadIdx.x >> 5;
  const int hl = lane >> 4;
  const int rl = lane & 15;
  const int wm = wave & 3;
  const int wn = wave >> 2;
  const int m0 = blockIdx.y * 128 + wm * 32;
  const int n0 = blockIdx.x * 128 + wn * 64;
  if (m0 >= M || n0 >= N) return;

  v8f acc[2][4];
  #pragma unroll
  for (int tm = 0; tm < 2; ++tm)
    #pragma unroll
    for (int tn = 0; tn < 4; ++tn) acc[tm][tn] = (v8f){};

  auto loadA = [&](v16h af[2], int kb) {
    #pragma unroll
    for (int tm = 0; tm < 2; ++tm) {
      const ft16* ap = A + (size_t)(m0 + tm * 16 + rl) * K + kb;
      af[tm] = cat16(*(const v8h*)(ap + hl * 8), *(const v8h*)(ap + 16 + hl * 8));
    }
  };
  auto loadB = [&](v16h bf[4], int kb) {
    #pragma unroll
    for (int tn = 0; tn < 4; ++tn) {
      const ft16* bp = BT + (size_t)(n0 + tn * 16 + rl) * K + kb + hl * 8;
      bf[tn] = cat16(*(const v8h*)bp, *(const v8h*)(bp + 16));
    }
  };
  auto mma = [&](const v16h af[2], const v16h bf[4]) {
    #pragma unroll
    for (int tm = 0; tm < 2; ++tm)
      #pragma unroll
      for (int tn = 0; tn < 4; ++tn)
        acc[tm][tn] = wmma_f16(af[tm], bf[tn], acc[tm][tn]);
  };

  v16h af[2], bf[4];
  loadA(af, 0);
  loadB(bf, 0);
  for (int kb = 0; kb + 32 < K; kb += 32) {
    v16h afn[2], bfn[4];
    loadA(afn, kb + 32);
    loadB(bfn, kb + 32);
    // warm L2 ahead on the streaming B side (global_prefetch_b8)
    __builtin_prefetch(BT + (size_t)(n0 + rl) * K + kb + 512, 0, 0);
    mma(af, bf);
    #pragma unroll
    for (int tm = 0; tm < 2; ++tm) af[tm] = afn[tm];
    #pragma unroll
    for (int tn = 0; tn < 4; ++tn) bf[tn] = bfn[tn];
  }
  mma(af, bf);

  #pragma unroll
  for (int tm = 0; tm < 2; ++tm) {
    #pragma unroll
    for (int tn = 0; tn < 4; ++tn) {
      #pragma unroll
      for (int r = 0; r < 8; ++r) {
        const int row = m0 + tm * 16 + r + 8 * hl;
        const int col = n0 + tn * 16 + rl;
        float v = acc[tm][tn][r];
        if (BIAS) v += bias[col];
        if (GELU) v = gelu_tanh(v);
        if (OUTMODE == 0) {
          float* Cf = (float*)Cout;
          const size_t idx = (size_t)row * N + col;
          Cf[idx] = ADD ? (Cf[idx] + v) : v;
        } else if (OUTMODE == 1) {
          ((ft16*)Cout)[(size_t)row * N + col] = (ft16)v;
        } else {
          // V-transposed: [B,H,DH,S]
          const int b = row >> 11;          // row / Sq
          const int s = row & (Sq - 1);
          const int h = col >> 8;           // col / DHq
          const int d = col & (DHq - 1);
          ((ft16*)Cout)[(((size_t)b * Hq + h) * DHq + d) * Sq + s] = (ft16)v;
        }
      }
    }
  }
}

// ---------------- Rotary (RD=64), in-place on f16 Q and K ---------------
__global__ __launch_bounds__(256)
void rotary_kernel(ft16* __restrict__ Qd, ft16* __restrict__ Kd) {
  const int idx = blockIdx.x * 256 + threadIdx.x;    // B*S*H*32 threads
  const int pair = idx & 31;
  const int h = (idx >> 5) & 15;
  const int s = (idx >> 9) & 2047;
  const int b = idx >> 20;
  const size_t base = (size_t)(b * Sq + s) * Dq + (size_t)h * DHq + 2 * pair;
  const float theta = (float)s * powf(10000.0f, -(float)(2 * pair) / 64.0f);
  const float cs = cosf(theta), sn = sinf(theta);
  float x1 = (float)Qd[base], x2 = (float)Qd[base + 1];
  Qd[base]     = (ft16)(x1 * cs - x2 * sn);
  Qd[base + 1] = (ft16)(x2 * cs + x1 * sn);
  x1 = (float)Kd[base]; x2 = (float)Kd[base + 1];
  Kd[base]     = (ft16)(x1 * cs - x2 * sn);
  Kd[base + 1] = (ft16)(x2 * cs + x1 * sn);
}

// ---------------- Flash attention: 1 wave = 16 queries x one head -------
// Q,K in [B,S,H*DH] f16; V pre-transposed [B,H,DH,S] f16.
__global__ __launch_bounds__(32)
void flash_attn(const ft16* __restrict__ Q, const ft16* __restrict__ Km,
                const ft16* __restrict__ VT, ft16* __restrict__ O) {
  const int lane = threadIdx.x & 31;
  const int hl = lane >> 4;
  const int rl = lane & 15;
  const int qi0 = blockIdx.x * 16;
  const int h = blockIdx.y;
  const int b = blockIdx.z;

  __shared__ __align__(16) ft16 lq[16 * DHq];
  __shared__ __align__(16) ft16 lp[16 * 32];

  const size_t headOff = (size_t)h * DHq;
  const ft16* qbase = Q  + (size_t)(b * Sq) * Dq + headOff;
  const ft16* kbase = Km + (size_t)(b * Sq) * Dq + headOff;
  const ft16* vbase = VT + ((size_t)(b * Hq + h)) * DHq * Sq;

  // Stage the 16x256 Q tile in LDS once.
  for (int r = 0; r < 16; ++r) {
    const uint32_t* src = (const uint32_t*)(qbase + (size_t)(qi0 + r) * Dq);
    uint32_t* dst = (uint32_t*)(lq + r * DHq);
    for (int x = lane; x < DHq / 2; x += 32) dst[x] = src[x];
  }
  __syncthreads();

  v8f acc[DHq / 16];
  #pragma unroll
  for (int d = 0; d < DHq / 16; ++d) acc[d] = (v8f){};
  float mi[8], li[8];
  #pragma unroll
  for (int r = 0; r < 8; ++r) { mi[r] = -1e30f; li[r] = 0.f; }
  const float scale = 0.0625f;  // 1/sqrt(256)

  for (int kb = 0; kb < qi0 + 16; kb += 32) {
    // ---- scores: two 16x16 tiles over a 32-key block (pipelined) ----
    v8f s0 = {}, s1 = {};
    auto loadK = [&](int kc, v16h& b0, v16h& b1) {
      const ft16* kp0 = kbase + (size_t)(kb + rl) * Dq + kc * 32 + hl * 8;
      b0 = cat16(*(const v8h*)kp0, *(const v8h*)(kp0 + 16));
      const ft16* kp1 = kp0 + (size_t)16 * Dq;
      b1 = cat16(*(const v8h*)kp1, *(const v8h*)(kp1 + 16));
    };
    v16h b0c, b1c;
    loadK(0, b0c, b1c);
    #pragma unroll
    for (int kc = 0; kc < DHq / 32; ++kc) {
      v16h b0n, b1n;
      if (kc < DHq / 32 - 1) loadK(kc + 1, b0n, b1n);
      const ft16* aq = lq + rl * DHq + kc * 32 + hl * 8;
      v16h a = cat16(*(const v8h*)aq, *(const v8h*)(aq + 16));
      s0 = wmma_f16(a, b0c, s0);
      s1 = wmma_f16(a, b1c, s1);
      if (kc < DHq / 32 - 1) { b0c = b0n; b1c = b1n; }
    }
    // ---- online softmax ----
    float pr0[8], pr1[8], alpha[8];
    #pragma unroll
    for (int r = 0; r < 8; ++r) {
      const int qrow = qi0 + r + 8 * hl;
      float v0 = s0[r] * scale; if (kb + rl > qrow)      v0 = -1e30f;
      float v1 = s1[r] * scale; if (kb + 16 + rl > qrow) v1 = -1e30f;
      float mx = fmaxf(v0, v1);
      mx = fmaxf(mx, __shfl_xor(mx, 1));
      mx = fmaxf(mx, __shfl_xor(mx, 2));
      mx = fmaxf(mx, __shfl_xor(mx, 4));
      mx = fmaxf(mx, __shfl_xor(mx, 8));
      const float mnew = fmaxf(mi[r], mx);
      const float al = __expf(mi[r] - mnew);
      const float p0 = __expf(v0 - mnew);
      const float p1 = __expf(v1 - mnew);
      float rs = p0 + p1;
      rs += __shfl_xor(rs, 1);
      rs += __shfl_xor(rs, 2);
      rs += __shfl_xor(rs, 4);
      rs += __shfl_xor(rs, 8);
      li[r] = li[r] * al + rs;
      mi[r] = mnew;
      alpha[r] = al;
      pr0[r] = p0; pr1[r] = p1;
    }
    #pragma unroll
    for (int d = 0; d < DHq / 16; ++d)
      #pragma unroll
      for (int r = 0; r < 8; ++r) acc[d][r] *= alpha[r];
    // ---- P (C-layout f32) -> LDS -> A-layout f16 fragment ----
    __syncthreads();
    #pragma unroll
    for (int r = 0; r < 8; ++r) {
      lp[(r + 8 * hl) * 32 + rl]      = (ft16)pr0[r];
      lp[(r + 8 * hl) * 32 + 16 + rl] = (ft16)pr1[r];
    }
    __syncthreads();
    const ft16* pp = lp + rl * 32 + hl * 8;
    v16h pa = cat16(*(const v8h*)pp, *(const v8h*)(pp + 16));
    // ---- P x V : contiguous b128 loads from [B,H,DH,S] V (pipelined) ----
    auto loadV = [&](int d) {
      const ft16* vp = vbase + (size_t)(d * 16 + rl) * Sq + kb + hl * 8;
      return cat16(*(const v8h*)vp, *(const v8h*)(vp + 16));
    };
    v16h vbc = loadV(0);
    #pragma unroll
    for (int d = 0; d < DHq / 16; ++d) {
      v16h vbn;
      if (d < DHq / 16 - 1) vbn = loadV(d + 1);
      acc[d] = wmma_f16(pa, vbc, acc[d]);
      if (d < DHq / 16 - 1) vbc = vbn;
    }
  }
  // ---- epilogue: normalize and write f16 attention output ----
  #pragma unroll
  for (int r = 0; r < 8; ++r) {
    const float inv = 1.0f / li[r];
    const size_t rowOff = (size_t)(b * Sq + qi0 + r + 8 * hl) * Dq + headOff;
    #pragma unroll
    for (int d = 0; d < DHq / 16; ++d)
      O[rowOff + d * 16 + rl] = (ft16)(acc[d][r] * inv);
  }
}

// ---------------- residual init ----------------------------------------
__global__ __launch_bounds__(256)
void copy_kernel(const float* __restrict__ src, float* __restrict__ dst, size_t n) {
  size_t i = (size_t)blockIdx.x * 256 + threadIdx.x;
  if (i < n) dst[i] = src[i];
}

extern "C" void kernel_launch(void* const* d_in, const int* in_sizes, int n_in,
                              void* d_out, int out_size, void* d_ws, size_t ws_size,
                              hipStream_t stream) {
  const float* hidden   = (const float*)d_in[0];
  const float* ln_w     = (const float*)d_in[1];
  const float* ln_b     = (const float*)d_in[2];
  const float* wq       = (const float*)d_in[3];
  const float* wk       = (const float*)d_in[4];
  const float* wv       = (const float*)d_in[5];
  const float* wo       = (const float*)d_in[6];
  const float* fc_in_w  = (const float*)d_in[7];
  const float* fc_in_b  = (const float*)d_in[8];
  const float* fc_out_w = (const float*)d_in[9];
  const float* fc_out_b = (const float*)d_in[10];
  float* out = (float*)d_out;

  const size_t BS  = (size_t)Bq * Sq;          // 4096 rows
  const size_t ACT = BS * Dq;                  // 16,777,216
  const size_t WDD = (size_t)Dq * Dq;          // 16,777,216
  const size_t WDF = (size_t)Dq * FFq;         // 67,108,864

  ft16* ln16   = (ft16*)d_ws;
  ft16* q16    = ln16 + ACT;
  ft16* k16    = q16 + ACT;
  ft16* vT16   = k16 + ACT;                    // [B,H,DH,S]
  ft16* attn16 = vT16 + ACT;
  ft16* h16    = attn16 + ACT;                 // BS * FF
  ft16* wqT    = h16 + (BS * FFq);
  ft16* wkT    = wqT + WDD;
  ft16* wvT    = wkT + WDD;
  ft16* woT    = wvT + WDD;
  ft16* fciT   = woT + WDD;                    // [FF, D]
  ft16* fcoT   = fciT + WDF;                   // [D, FF]

  // 0) One-time weight transpose+convert to f16 [N,K]
  dim3 tDD(Dq / 32, Dq / 32);
  transpose_cvt<<<tDD, 256, 0, stream>>>(wq, wqT, Dq, Dq);
  transpose_cvt<<<tDD, 256, 0, stream>>>(wk, wkT, Dq, Dq);
  transpose_cvt<<<tDD, 256, 0, stream>>>(wv, wvT, Dq, Dq);
  transpose_cvt<<<tDD, 256, 0, stream>>>(wo, woT, Dq, Dq);
  transpose_cvt<<<dim3(FFq / 32, Dq / 32), 256, 0, stream>>>(fc_in_w, fciT, Dq, FFq);
  transpose_cvt<<<dim3(Dq / 32, FFq / 32), 256, 0, stream>>>(fc_out_w, fcoT, FFq, Dq);

  // 1) LayerNorm -> f16
  ln_kernel<<<(int)BS, 256, 0, stream>>>(hidden, ln_w, ln_b, ln16);

  // 2) Q/K/V projections
  dim3 gD(Dq / 128, (int)(BS / 128));
  gemm_wmma<false,false,false,1><<<gD, 256, 0, stream>>>(ln16, wqT, nullptr, q16, (int)BS, Dq, Dq);
  gemm_wmma<false,false,false,1><<<gD, 256, 0, stream>>>(ln16, wkT, nullptr, k16, (int)BS, Dq, Dq);
  gemm_wmma<false,false,false,2><<<gD, 256, 0, stream>>>(ln16, wvT, nullptr, vT16, (int)BS, Dq, Dq);

  // 3) Rotary embedding on Q, K (in place)
  rotary_kernel<<<(Bq * Sq * Hq * 32) / 256, 256, 0, stream>>>(q16, k16);

  // 4) Causal flash attention
  flash_attn<<<dim3(Sq / 16, Hq, Bq), 32, 0, stream>>>(q16, k16, vT16, attn16);

  // 5) out = residual
  copy_kernel<<<(int)(ACT / 256), 256, 0, stream>>>(hidden, out, ACT);

  // 6) out += attn @ wo
  gemm_wmma<true,false,false,0><<<gD, 256, 0, stream>>>(attn16, woT, nullptr, out, (int)BS, Dq, Dq);

  // 7) h = gelu(ln @ fc_in_w + fc_in_b) -> f16
  gemm_wmma<false,true,true,1><<<dim3(FFq / 128, (int)(BS / 128)), 256, 0, stream>>>(
      ln16, fciT, fc_in_b, h16, (int)BS, FFq, Dq);

  // 8) out += h @ fc_out_w + fc_out_b
  gemm_wmma<true,false,true,0><<<gD, 256, 0, stream>>>(h16, fcoT, fc_out_b, out, (int)BS, Dq, FFq);
}
